// WildfireMamba_7155415515365
// MI455X (gfx1250) — matrix operand order repaired and checked by reading, not compiled, not run on hardware
//
#include <hip/hip_runtime.h>
#include <hip/hip_bf16.h>
#include <math.h>

// ---------------------------------------------------------------------------
// WildfireMamba on MI455X (gfx1250).
// All big matmuls use V_WMMA_F32_16X16X32_F16 (f16 operands, f32 accumulate).
// Generic GEMM stages its 16x32 A-tile through LDS with
// GLOBAL_LOAD_ASYNC_TO_LDS_B128 (ASYNCcnt-tracked), one workgroup per 16-row
// M-stripe with one wave per 16-column N-tile sharing the A fragment.
// The SSM scan keeps state in WMMA accumulators, round-trips it through a
// 4KB LDS tile each step; AB B-fragments are register-resident for all T=64.
// ---------------------------------------------------------------------------

typedef _Float16 h16;
typedef __attribute__((ext_vector_type(16))) _Float16 v16h;
typedef __attribute__((ext_vector_type(8)))  float    v8f;

#if defined(__AMDGCN__) && __has_builtin(__builtin_amdgcn_global_load_async_to_lds_b128)
#define HAVE_ASYNC_LDS 1
typedef int v4i_ __attribute__((vector_size(16)));
typedef __attribute__((address_space(1))) v4i_* gbl_v4i;
typedef __attribute__((address_space(3))) v4i_* lds_v4i;
#else
#define HAVE_ASYNC_LDS 0
#endif

#define B__ 4
#define T__ 64
#define N__ 1000
#define F__ 24
#define H__ 128
#define S__ 64
#define KW__ 5
#define GH__ 64
#define BN__ (B__ * N__)          // 4000
#define M0__ (BN__ * T__)         // 256000
#define FP__ 32                   // F padded to K-multiple of 32
#define NKP__ 1024                // N padded for K dimension (mult of 32)
#define NMP__ 1008                // N padded for M dimension (mult of 16)

// ---- WMMA fragment builders (gfx1250 wave32 layouts, ISA 7.12.2) ----------

// A: 16x32 f16, row-major source with leading dim lda.
// lane 0-15: row=lane, K 0..7 (elems 0-7) and 16..23 (elems 8-15)
// lane 16-31: row=lane-16, K 8..15 and 24..31
__device__ __forceinline__ v16h load_a_frag(const h16* A, long lda, int lane) {
  const int row = lane & 15;
  const int kb  = (lane >> 4) * 8;
  const h16* p = A + (long)row * lda;
  v16h a;
#pragma unroll
  for (int i = 0; i < 8; ++i) {
    a[i]     = p[kb + i];
    a[8 + i] = p[16 + kb + i];
  }
  return a;
}

// B: 32x16 f16, row-major source (K x N) with leading dim ldb.
// lane 0-15: col=lane, K 0..7 (elems 0-7) and 16..23 (elems 8-15)
// lane 16-31: col=lane-16, K 8..15 and 24..31
__device__ __forceinline__ v16h load_b_frag(const h16* Bm, long ldb, int lane) {
  const int n  = lane & 15;
  const int kb = (lane >> 4) * 8;
  const h16* p = Bm + n;
  v16h b;
#pragma unroll
  for (int j = 0; j < 8; ++j) {
    b[j]     = p[(long)(kb + j) * ldb];
    b[8 + j] = p[(long)(16 + kb + j) * ldb];
  }
  return b;
}

__device__ __forceinline__ v8f wmma_f16(v16h a, v16h b, v8f c) {
  return __builtin_amdgcn_wmma_f32_16x16x32_f16(false, a, false, b,
                                                (short)0, c, false, false);
}

// ---- generic batched GEMM: O = act(A(f16) @ W(f16) + bias) ----------------
// grid = (Mtiles, 1, batches), block = 32 * (Ncols/16) threads.
// Workgroup covers one 16-row M-stripe across all N columns; 16x32 A-tile is
// staged in LDS (async global->LDS when available) and shared by all waves.
// act: 0=none 1=sigmoid 2=relu
__global__ void gemm_f16_kernel(const h16* __restrict__ A, long lda, long sAb,
                                const h16* __restrict__ W, long ldw, long sWb,
                                const float* __restrict__ bias,
                                h16* __restrict__ O, long ldo, long sOb,
                                int ktiles, int act) {
  __shared__ __align__(16) h16 atile[16 * 32];
  const int tid = threadIdx.x;
  const int wave = tid >> 5, lane = tid & 31;
  const int n0 = wave * 16;
  const long m0 = (long)blockIdx.x * 16;
  const int bz = blockIdx.z;
  const h16* Ab = A + (long)bz * sAb + m0 * lda;
  const h16* Wb = W + (long)bz * sWb + n0;
  h16* Ob = O + (long)bz * sOb;

  v8f acc = {};
  for (int kc = 0; kc < ktiles; ++kc) {
    // stage A tile: 16 rows x 32 halves (64B/row) -> 64 lanes x 16B
    if (tid < 64) {
      const int row = tid >> 2, seg = tid & 3;
      const h16* gsrc = Ab + (long)kc * 32 + (long)row * lda + seg * 8;
      h16* ldst = atile + row * 32 + seg * 8;
#if HAVE_ASYNC_LDS
      __builtin_amdgcn_global_load_async_to_lds_b128(
          (gbl_v4i)(h16*)gsrc, (lds_v4i)ldst, 0, 0);
#else
      *(float4*)ldst = *(const float4*)gsrc;
#endif
    }
#if HAVE_ASYNC_LDS
#if __has_builtin(__builtin_amdgcn_s_wait_asynccnt)
    __builtin_amdgcn_s_wait_asynccnt(0);
#else
    asm volatile("s_wait_asynccnt 0" ::: "memory");
#endif
#endif
    __syncthreads();
    v16h af = load_a_frag(atile, 32, lane);
    v16h bf = load_b_frag(Wb + (long)kc * 32 * ldw, ldw, lane);
    acc = wmma_f16(af, bf, acc);
    __syncthreads();
  }
  const int col = lane & 15, hi = lane >> 4;
#pragma unroll
  for (int r = 0; r < 8; ++r) {
    const int row = r + hi * 8;
    const int c = n0 + col;
    float v = acc[r] + (bias ? bias[c] : 0.0f);
    if (act == 1) v = 1.0f / (1.0f + expf(-v));
    else if (act == 2) v = fmaxf(v, 0.0f);
    Ob[(m0 + row) * ldo + c] = (h16)v;
  }
}

// ---- input prep: x(B,T,N,F) -> xpad/deltapad (BN,T,FP) f16 ----------------
__global__ void prep_kernel(const float* __restrict__ x,
                            h16* __restrict__ xp, h16* __restrict__ dp) {
  long i = (long)blockIdx.x * blockDim.x + threadIdx.x; // i = bn*T + t
  if (i >= (long)M0__) return;
  const int t = (int)(i % T__);
  const long bn = i / T__;
  const int n = (int)(bn % N__);
  const int b = (int)(bn / N__);
  const float* src = x + (((long)b * T__ + t) * N__ + n) * F__;
  const float* sp  = x + (((long)b * T__ + (t - 1)) * N__ + n) * F__;
  h16* xo = xp + i * FP__;
  h16* dd = dp + i * FP__;
#pragma unroll
  for (int f = 0; f < F__; ++f) {
    const float v = src[f];
    xo[f] = (h16)v;
    dd[f] = (h16)(t > 0 ? (v - sp[f]) : 0.0f);
  }
#pragma unroll
  for (int f = F__; f < FP__; ++f) { xo[f] = (h16)0.f; dd[f] = (h16)0.f; }
}

// ---- depthwise conv along T (K=5, pad=2) ----------------------------------
__global__ void dwconv_kernel(const h16* __restrict__ h,
                              const float* __restrict__ w,
                              const float* __restrict__ b,
                              h16* __restrict__ hc) {
  long i = (long)blockIdx.x * blockDim.x + threadIdx.x;
  if (i >= (long)M0__ * H__) return;
  const int ch = (int)(i % H__);
  const long mt = i / H__;            // bn*T + t
  const int t = (int)(mt % T__);
  const long base = (mt - t) * H__ + ch;
  float acc = b[ch];
#pragma unroll
  for (int k = 0; k < KW__; ++k) {
    const int tt = t + k - 2;
    if (tt >= 0 && tt < T__)
      acc += (float)h[base + (long)tt * H__] * w[ch * KW__ + k];
  }
  hc[i] = (h16)acc;
}

// ---- AB = A(H,S) @ Bm(S,H) in f32, stored f16 -----------------------------
__global__ void ab_kernel(const float* __restrict__ A,
                          const float* __restrict__ Bm, h16* __restrict__ AB) {
  const int i = blockIdx.x * blockDim.x + threadIdx.x;
  if (i >= H__ * H__) return;
  const int r = i / H__, c = i % H__;
  float s = 0.f;
#pragma unroll 8
  for (int k = 0; k < S__; ++k) s += A[r * S__ + k] * Bm[k * H__ + c];
  AB[i] = (h16)s;
}

// ---- gated SSM scan: 16 sequences per workgroup, 8 waves x 16 cols --------
__global__ void scan_kernel(const h16* __restrict__ g, const h16* __restrict__ hc,
                            const h16* __restrict__ AB, h16* __restrict__ ys) {
  __shared__ h16 st_lds[16 * H__];
  const int tid = threadIdx.x;
  const int wave = tid >> 5, lane = tid & 31;
  const int n0 = wave * 16;
  const int col = lane & 15, hi = lane >> 4;
  const long bn0 = (long)blockIdx.x * 16;

  // AB B-fragments, register resident for all 64 steps
  v16h bf[4];
#pragma unroll
  for (int kc = 0; kc < 4; ++kc)
    bf[kc] = load_b_frag(AB + (long)kc * 32 * H__ + n0, H__, lane);

  v8f st = {};  // f32 state tile (16 rows x 16 cols of this wave)
  for (int t = 0; t < T__; ++t) {
#pragma unroll
    for (int r = 0; r < 8; ++r)
      st_lds[(r + hi * 8) * H__ + n0 + col] = (h16)st[r];
    __syncthreads();
    v16h af[4];
#pragma unroll
    for (int kc = 0; kc < 4; ++kc)
      af[kc] = load_a_frag(st_lds + kc * 32, H__, lane);
    __syncthreads();  // reads complete; next iter may overwrite LDS
    v8f acc = {};
#pragma unroll
    for (int kc = 0; kc < 4; ++kc)
      acc = wmma_f16(af[kc], bf[kc], acc);
#pragma unroll
    for (int r = 0; r < 8; ++r) {
      const int row = r + hi * 8;
      const long idx = ((bn0 + row) * T__ + t) * (long)H__ + n0 + col;
      const float gv = (float)g[idx];
      const float hv = (float)hc[idx];
      const float ns = gv * (acc[r] + hv) + (1.0f - gv) * st[r];
      st[r] = ns;
      ys[idx] = (h16)ns;
    }
    if (t + 1 < T__) {
      const long nidx = ((bn0 + hi * 8) * T__ + t + 1) * (long)H__ + n0 + col;
      __builtin_prefetch(g + nidx, 0, 1);   // global_prefetch_b8
      __builtin_prefetch(hc + nidx, 0, 1);
    }
  }
}

// ---- fused out-proj + residual + LayerNorm --------------------------------
// grid = M0/16 blocks, 256 threads = 8 waves covering all 128 output columns
__global__ void outproj_ln_kernel(const h16* __restrict__ ys, const h16* __restrict__ hc,
                                  const h16* __restrict__ ow, const float* __restrict__ ob,
                                  const float* __restrict__ lg, const float* __restrict__ lb,
                                  h16* __restrict__ yo) {
  __shared__ float ybuf[16 * H__];
  __shared__ float mu_s[16], rs_s[16];
  const int tid = threadIdx.x;
  const int wave = tid >> 5, lane = tid & 31;
  const int n0 = wave * 16;
  const long m0 = (long)blockIdx.x * 16;

  v8f acc = {};
#pragma unroll
  for (int kc = 0; kc < 4; ++kc) {
    v16h af = load_a_frag(ys + m0 * H__ + (long)kc * 32, H__, lane);
    v16h bfr = load_b_frag(ow + (long)kc * 32 * H__ + n0, H__, lane);
    acc = wmma_f16(af, bfr, acc);
  }
  const int col = lane & 15, hi = lane >> 4;
#pragma unroll
  for (int r = 0; r < 8; ++r) {
    const int row = r + hi * 8;
    const int c = n0 + col;
    ybuf[row * H__ + c] = acc[r] + ob[c] + (float)hc[(m0 + row) * H__ + c];
  }
  __syncthreads();
  if (tid < 16) {
    float mu = 0.f;
    for (int c2 = 0; c2 < H__; ++c2) mu += ybuf[tid * H__ + c2];
    mu *= (1.0f / H__);
    float var = 0.f;
    for (int c2 = 0; c2 < H__; ++c2) {
      const float d = ybuf[tid * H__ + c2] - mu;
      var += d * d;
    }
    mu_s[tid] = mu;
    rs_s[tid] = rsqrtf(var * (1.0f / H__) + 1e-5f);
  }
  __syncthreads();
  {
    const int row = tid >> 4;
    const int cb = (tid & 15) * 8;
    const float mu = mu_s[row], rs = rs_s[row];
#pragma unroll
    for (int i = 0; i < 8; ++i) {
      const int c = cb + i;
      const float v = (ybuf[row * H__ + c] - mu) * rs * lg[c] + lb[c];
      yo[(m0 + row) * H__ + c] = (h16)v;
    }
  }
}

// ---- gate = sigmoid(delta@dg_w+b); fused = base*gate + delta (padded) -----
__global__ void gate_fuse_kernel(const h16* __restrict__ dy, const h16* __restrict__ by,
                                 const h16* __restrict__ dgw, const float* __restrict__ dgb,
                                 h16* __restrict__ fused) {
  const int lane = threadIdx.x;
  const long m0 = (long)blockIdx.x * 16;
  const int n0 = blockIdx.y * 16;
  const long lda = (long)T__ * H__;            // row bn -> last timestep
  const h16* A = dy + (long)(T__ - 1) * H__ + m0 * lda;

  v8f acc = {};
#pragma unroll
  for (int kc = 0; kc < 4; ++kc) {
    v16h af = load_a_frag(A + (long)kc * 32, lda, lane);
    v16h bfr = load_b_frag(dgw + (long)kc * 32 * H__ + n0, H__, lane);
    acc = wmma_f16(af, bfr, acc);
  }
  const int col = lane & 15, hi = lane >> 4;
#pragma unroll
  for (int r = 0; r < 8; ++r) {
    const int row = r + hi * 8;
    const long m = m0 + row;
    const int c = n0 + col;
    const float gate = 1.0f / (1.0f + expf(-(acc[r] + dgb[c])));
    const long src = m * lda + (long)(T__ - 1) * H__ + c;
    const float fv = (float)by[src] * gate + (float)dy[src];
    const int b = (int)(m / N__), n = (int)(m % N__);
    fused[((long)b * NKP__ + n) * H__ + c] = (h16)fv;
  }
}

// ---- adjacency normalize: (adj+I)/rowsum, f16 padded (NMP x NKP) ----------
__global__ void adjnorm_kernel(const float* __restrict__ adj, h16* __restrict__ adjn) {
  __shared__ float red[256];
  const int i = blockIdx.x;   // row 0..999
  const int tid = threadIdx.x;
  float s = 0.f;
  for (int j = tid; j < N__; j += 256) s += adj[(long)i * N__ + j];
  red[tid] = s;
  __syncthreads();
  for (int off = 128; off > 0; off >>= 1) {
    if (tid < off) red[tid] += red[tid + off];
    __syncthreads();
  }
  const float inv = 1.0f / fmaxf(red[0] + 1.0f, 1e-6f);
  for (int j = tid; j < N__; j += 256) {
    const float v = adj[(long)i * N__ + j] + (j == i ? 1.0f : 0.0f);
    adjn[(long)i * NKP__ + j] = (h16)(v * inv);
  }
}

// ---- final classifier: logits = spatial @ cls_w + cls_b -------------------
__global__ void logits_kernel(const h16* __restrict__ sp, const float* __restrict__ cw,
                              const float* __restrict__ cb, float* __restrict__ out) {
  const int i = blockIdx.x * blockDim.x + threadIdx.x;
  if (i >= BN__) return;
  const int b = i / N__, n = i % N__;
  const h16* p = sp + ((long)b * NMP__ + n) * GH__;
  float s = cb[0];
#pragma unroll 8
  for (int k = 0; k < GH__; ++k) s += (float)p[k] * cw[k];
  out[i] = s;
}

// ---- small utilities ------------------------------------------------------
__global__ void cvt_f16_kernel(const float* __restrict__ s, h16* __restrict__ d, long n) {
  const long i = (long)blockIdx.x * blockDim.x + threadIdx.x;
  if (i < n) d[i] = (h16)s[i];
}
__global__ void cvt_pad_kernel(const float* __restrict__ s, h16* __restrict__ d,
                               int K, int Kp, int N) {
  const long i = (long)blockIdx.x * blockDim.x + threadIdx.x;
  if (i >= (long)Kp * N) return;
  const int k = (int)(i / N);
  d[i] = (k < K) ? (h16)s[i] : (h16)0.f;
}
__global__ void fill_f16_kernel(h16* __restrict__ p, long n) {
  const long i = (long)blockIdx.x * blockDim.x + threadIdx.x;
  if (i < n) p[i] = (h16)0.f;
}

// ---------------------------------------------------------------------------
extern "C" void kernel_launch(void* const* d_in, const int* in_sizes, int n_in,
                              void* d_out, int out_size, void* d_ws, size_t ws_size,
                              hipStream_t stream) {
  (void)in_sizes; (void)out_size; (void)ws_size;
  if (n_in < 46) return;

  const float* x   = (const float*)d_in[0];
  const float* adj = (const float*)d_in[1];
  // params: jax pytree flatten = sorted dict keys, lists in order
  const float* cls_b = (const float*)d_in[2];
  const float* cls_w = (const float*)d_in[3];
  struct Blk { const float *A,*Bm,*dw_b,*dw_w,*gate_b,*gate_w,*in_b,*in_w,*ln_b,*ln_g,*out_b,*out_w; };
  auto get_blk = [&](int base) {
    Blk k;
    k.A = (const float*)d_in[base + 0];  k.Bm = (const float*)d_in[base + 1];
    k.dw_b = (const float*)d_in[base + 2]; k.dw_w = (const float*)d_in[base + 3];
    k.gate_b = (const float*)d_in[base + 4]; k.gate_w = (const float*)d_in[base + 5];
    k.in_b = (const float*)d_in[base + 6];  k.in_w = (const float*)d_in[base + 7];
    k.ln_b = (const float*)d_in[base + 8];  k.ln_g = (const float*)d_in[base + 9];
    k.out_b = (const float*)d_in[base + 10]; k.out_w = (const float*)d_in[base + 11];
    return k;
  };
  Blk dblk = get_blk(4);
  const float* dg_b   = (const float*)d_in[16];
  const float* dg_w   = (const float*)d_in[17];
  const float* gcn_b1 = (const float*)d_in[18];
  const float* gcn_b2 = (const float*)d_in[19];
  const float* gcn_w1 = (const float*)d_in[20];
  const float* gcn_w2 = (const float*)d_in[21];
  Blk tblk0 = get_blk(22);
  Blk tblk1 = get_blk(34);

  // workspace carve
  char* ws = (char*)d_ws;
  size_t off = 0;
  auto alloc = [&](size_t bytes) -> void* {
    off = (off + 255) & ~(size_t)255;
    void* p = ws + off; off += bytes; return p;
  };
  h16* xpad = (h16*)alloc((size_t)M0__ * FP__ * 2);
  h16* dpad = (h16*)alloc((size_t)M0__ * FP__ * 2);
  h16* ioA  = (h16*)alloc((size_t)M0__ * H__ * 2);
  h16* ioB  = (h16*)alloc((size_t)M0__ * H__ * 2);
  h16* hbuf = (h16*)alloc((size_t)M0__ * H__ * 2);
  h16* hcb  = (h16*)alloc((size_t)M0__ * H__ * 2);
  h16* gbuf = (h16*)alloc((size_t)M0__ * H__ * 2);
  h16* ysb  = (h16*)alloc((size_t)M0__ * H__ * 2);
  h16* w_in0  = (h16*)alloc(FP__ * H__ * 2);
  h16* w_in1  = (h16*)alloc(H__ * H__ * 2);
  h16* w_ind  = (h16*)alloc(FP__ * H__ * 2);
  h16* w_gt0  = (h16*)alloc(H__ * H__ * 2);
  h16* w_gt1  = (h16*)alloc(H__ * H__ * 2);
  h16* w_gtd  = (h16*)alloc(H__ * H__ * 2);
  h16* w_ot0  = (h16*)alloc(H__ * H__ * 2);
  h16* w_ot1  = (h16*)alloc(H__ * H__ * 2);
  h16* w_otd  = (h16*)alloc(H__ * H__ * 2);
  h16* ab0    = (h16*)alloc(H__ * H__ * 2);
  h16* ab1    = (h16*)alloc(H__ * H__ * 2);
  h16* abd    = (h16*)alloc(H__ * H__ * 2);
  h16* w_dg   = (h16*)alloc(H__ * H__ * 2);
  h16* w_g1   = (h16*)alloc(H__ * GH__ * 2);
  h16* w_g2   = (h16*)alloc(GH__ * GH__ * 2);
  h16* adjn   = (h16*)alloc((size_t)NMP__ * NKP__ * 2);
  h16* fused  = (h16*)alloc((size_t)B__ * NKP__ * H__ * 2);
  h16* z1     = (h16*)alloc((size_t)B__ * NMP__ * H__ * 2);
  h16* z2     = (h16*)alloc((size_t)B__ * NKP__ * GH__ * 2);
  h16* z3     = (h16*)alloc((size_t)B__ * NMP__ * GH__ * 2);
  h16* spat   = (h16*)alloc((size_t)B__ * NMP__ * GH__ * 2);

  auto blks = [](long n) { return (unsigned)((n + 255) / 256); };

  // weight conversions (f32 -> f16, K-padded where needed)
  cvt_pad_kernel<<<blks((long)FP__ * H__), 256, 0, stream>>>(tblk0.in_w, w_in0, F__, FP__, H__);
  cvt_pad_kernel<<<blks((long)FP__ * H__), 256, 0, stream>>>(dblk.in_w,  w_ind, F__, FP__, H__);
  cvt_f16_kernel<<<blks((long)H__ * H__), 256, 0, stream>>>(tblk1.in_w, w_in1, (long)H__ * H__);
  cvt_f16_kernel<<<blks((long)H__ * H__), 256, 0, stream>>>(tblk0.gate_w, w_gt0, (long)H__ * H__);
  cvt_f16_kernel<<<blks((long)H__ * H__), 256, 0, stream>>>(tblk1.gate_w, w_gt1, (long)H__ * H__);
  cvt_f16_kernel<<<blks((long)H__ * H__), 256, 0, stream>>>(dblk.gate_w,  w_gtd, (long)H__ * H__);
  cvt_f16_kernel<<<blks((long)H__ * H__), 256, 0, stream>>>(tblk0.out_w, w_ot0, (long)H__ * H__);
  cvt_f16_kernel<<<blks((long)H__ * H__), 256, 0, stream>>>(tblk1.out_w, w_ot1, (long)H__ * H__);
  cvt_f16_kernel<<<blks((long)H__ * H__), 256, 0, stream>>>(dblk.out_w,  w_otd, (long)H__ * H__);
  cvt_f16_kernel<<<blks((long)H__ * H__), 256, 0, stream>>>(dg_w, w_dg, (long)H__ * H__);
  cvt_f16_kernel<<<blks((long)H__ * GH__), 256, 0, stream>>>(gcn_w1, w_g1, (long)H__ * GH__);
  cvt_f16_kernel<<<blks((long)GH__ * GH__), 256, 0, stream>>>(gcn_w2, w_g2, (long)GH__ * GH__);
  ab_kernel<<<blks((long)H__ * H__), 256, 0, stream>>>(tblk0.A, tblk0.Bm, ab0);
  ab_kernel<<<blks((long)H__ * H__), 256, 0, stream>>>(tblk1.A, tblk1.Bm, ab1);
  ab_kernel<<<blks((long)H__ * H__), 256, 0, stream>>>(dblk.A,  dblk.Bm,  abd);

  // zero-fill padded buffers (ws is poisoned between runs)
  fill_f16_kernel<<<blks((long)NMP__ * NKP__), 256, 0, stream>>>(adjn, (long)NMP__ * NKP__);
  fill_f16_kernel<<<blks((long)B__ * NKP__ * H__), 256, 0, stream>>>(fused, (long)B__ * NKP__ * H__);
  fill_f16_kernel<<<blks((long)B__ * NKP__ * GH__), 256, 0, stream>>>(z2, (long)B__ * NKP__ * GH__);

  prep_kernel<<<blks((long)M0__), 256, 0, stream>>>(x, xpad, dpad);
  adjnorm_kernel<<<N__, 256, 0, stream>>>(adj, adjn);

  const dim3 gemm_full(M0__ / 16, 1, 1);   // 256 threads: 8 waves x 16 cols
  auto run_block = [&](const Blk& bk, const h16* in, int inKtiles, const h16* w_in,
                       const h16* w_gate, const h16* w_out, const h16* w_ab, h16* out) {
    const long inLda = (long)inKtiles * 32;
    // h = in @ in_w + in_b
    gemm_f16_kernel<<<gemm_full, 256, 0, stream>>>(in, inLda, 0, w_in, H__, 0,
                                                   bk.in_b, hbuf, H__, 0, inKtiles, 0);
    // hc = dwconv(h)
    dwconv_kernel<<<blks((long)M0__ * H__), 256, 0, stream>>>(hbuf, bk.dw_w, bk.dw_b, hcb);
    // g = sigmoid(hc @ gate_w + gate_b)
    gemm_f16_kernel<<<gemm_full, 256, 0, stream>>>(hcb, H__, 0, w_gate, H__, 0,
                                                   bk.gate_b, gbuf, H__, 0, H__ / 32, 1);
    // gated scan over T
    scan_kernel<<<BN__ / 16, 256, 0, stream>>>(gbuf, hcb, w_ab, ysb);
    // y = LN(ys @ out_w + out_b + hc)
    outproj_ln_kernel<<<M0__ / 16, 256, 0, stream>>>(ysb, hcb, w_out, bk.out_b,
                                                     bk.ln_g, bk.ln_b, out);
  };

  run_block(tblk0, xpad, 1, w_in0, w_gt0, w_ot0, ab0, ioA);  // temporal[0]
  run_block(tblk1, ioA,  4, w_in1, w_gt1, w_ot1, ab1, ioB);  // temporal[1] -> base
  run_block(dblk,  dpad, 1, w_ind, w_gtd, w_otd, abd, ioA);  // delta       -> delta

  // fused = base * sigmoid(delta@dg_w+b) + delta, written into padded layout
  gate_fuse_kernel<<<dim3(BN__ / 16, H__ / 16, 1), 32, 0, stream>>>(ioA, ioB, w_dg, dg_b, fused);

  // GCN: z1 = adjn @ fused ; z2 = relu(z1@W1+b1) ; z3 = adjn @ z2 ; spat = relu(z3@W2+b2)
  gemm_f16_kernel<<<dim3(NMP__ / 16, 1, B__), 256, 0, stream>>>(
      adjn, NKP__, 0, fused, H__, (long)NKP__ * H__, nullptr,
      z1, H__, (long)NMP__ * H__, NKP__ / 32, 0);
  gemm_f16_kernel<<<dim3(NMP__ / 16, 1, B__), 128, 0, stream>>>(
      z1, H__, (long)NMP__ * H__, w_g1, GH__, 0, gcn_b1,
      z2, GH__, (long)NKP__ * GH__, H__ / 32, 2);
  gemm_f16_kernel<<<dim3(NMP__ / 16, 1, B__), 128, 0, stream>>>(
      adjn, NKP__, 0, z2, GH__, (long)NKP__ * GH__, nullptr,
      z3, GH__, (long)NMP__ * GH__, NKP__ / 32, 0);
  gemm_f16_kernel<<<dim3(NMP__ / 16, 1, B__), 128, 0, stream>>>(
      z3, GH__, (long)NMP__ * GH__, w_g2, GH__, 0, gcn_b2,
      spat, GH__, (long)NMP__ * GH__, GH__ / 32, 2);

  logits_kernel<<<blks((long)BN__), 256, 0, stream>>>(spat, cls_w, cls_b, (float*)d_out);
}
